// SRVQ3_38242388804096
// MI455X (gfx1250) — matrix operand order repaired and checked by reading, not compiled, not run on hardware
//
#include <hip/hip_runtime.h>
#include <hip/hip_bf16.h>
#include <math.h>

typedef __bf16 bf16;
typedef __attribute__((ext_vector_type(16))) __bf16 v16bf;
typedef __attribute__((ext_vector_type(8)))  __bf16 v8bf;
typedef __attribute__((ext_vector_type(8)))  float  v8f;

#define BN_EPS 1e-5f

// ---------------------------------------------------------------------------
// Zero the t=-1 pad row of a padded channels-last activation buffer
// layout [b][1+L][Ci]; pad row is row 0 of each batch image.
// ---------------------------------------------------------------------------
__global__ __launch_bounds__(256)
void zero_pad(bf16* __restrict__ buf, int Ci, int strideElems)
{
    int i = blockIdx.x * blockDim.x + threadIdx.x;
    if (i >= 32 * Ci) return;
    int b = i / Ci, c = i - b * Ci;
    buf[(size_t)b * strideElems + c] = (bf16)0.f;
}

// ---------------------------------------------------------------------------
// Layer 1: C_in = 1 -> plain VALU, fused BN+ReLU, padded channels-last out
// out layout: [b][1+t][co=32], co fastest -> coalesced u16 stores
// ---------------------------------------------------------------------------
__global__ __launch_bounds__(256)
void conv1_kernel(const float* __restrict__ x, const float* __restrict__ w,
                  const float* __restrict__ gam, const float* __restrict__ bet,
                  const float* __restrict__ mu,  const float* __restrict__ var,
                  bf16* __restrict__ out, int Lin, int Lout)
{
    int idx = blockIdx.x * blockDim.x + threadIdx.x;
    int total = 32 * 32 * Lout;
    if (idx >= total) return;
    int co = idx & 31;
    int t  = (idx >> 5) % Lout;
    int b  = idx / (Lout * 32);
    const float* xb = x + (size_t)b * Lin;
    int ti = 2 * t - 1;
    float acc = 0.f;
    float w0 = w[co*3+0], w1 = w[co*3+1], w2 = w[co*3+2];
    if (ti >= 0)      acc += w0 * xb[ti];
    acc += w1 * xb[ti + 1];
    if (ti + 2 < Lin) acc += w2 * xb[ti + 2];
    float s = gam[co] * rsqrtf(var[co] + BN_EPS);
    float v = (acc - mu[co]) * s + bet[co];
    out[((size_t)b * (Lout + 1) + 1 + t) * 32 + co] = (bf16)fmaxf(v, 0.f);
}

// ---------------------------------------------------------------------------
// Weight pre-swizzle for conv layers 2..6: frag-ready bf16.
// Wf[co_tile][kc][lane][j] = w[co][ci][kk], with K-order k = kk*Ci + ci and
// the contiguous A-fragment order (k = kc*32 + 16*grp + j, as the B loads use) baked in.
// ---------------------------------------------------------------------------
__global__ __launch_bounds__(256)
void prep_convw(const float* __restrict__ w, bf16* __restrict__ wf,
                int Ci, int Co, int ciShift, int nk)
{
    int i = blockIdx.x * blockDim.x + threadIdx.x;
    int total = Co * 3 * Ci;
    if (i >= total) return;
    int j    = i & 15;
    int lane = (i >> 4) & 31;
    int rem  = i >> 9;
    int kc   = rem % nk;
    int cot  = rem / nk;
    int grp = lane >> 4, mn = lane & 15;
    int p = j >> 1, h = j & 1;
    int ka = grp * 16 + 2 * p + h;
    int kglob = kc * 32 + ka;                 // k = kk*Ci + ci
    int kk = kglob >> ciShift;
    int ci = kglob & (Ci - 1);
    int co = cot * 16 + mn;
    wf[i] = (bf16)w[(size_t)co * (3 * Ci) + ci * 3 + kk];
}

// ---------------------------------------------------------------------------
// Layers 2..6: implicit-GEMM conv via V_WMMA_F32_16X16X32_BF16.
// One wave computes ALL Co/16 co-tiles for its 16-wide t-tile: per K-chunk,
// the activation B-fragment is loaded once (unconditional, pad row supplies
// the t=-1 zeros) and reused for Co/16 WMMAs. All strides compile-time.
// ---------------------------------------------------------------------------
template<int CI, int CO, int LIN, int LOUT>
__global__ __launch_bounds__(32)
void convw_kernel(const bf16* __restrict__ in, const bf16* __restrict__ wf,
                  const float* __restrict__ gam, const float* __restrict__ bet,
                  const float* __restrict__ mu,  const float* __restrict__ var,
                  bf16* __restrict__ out)
{
    constexpr int NK      = (3 * CI) >> 5;
    constexpr int NCOT    = CO / 16;
    constexpr int CISHIFT = (CI == 32) ? 5 : (CI == 64) ? 6 : 7;
    int lane = threadIdx.x;
    int grp  = lane >> 4;
    int mn   = lane & 15;
    int t0   = blockIdx.x * 16;
    int b    = blockIdx.y;
    const bf16* inb = in + ((size_t)b * (LIN + 1) + 1) * CI;  // skip pad row
    int t = t0 + mn;

    v8f acc[NCOT] = {};
    for (int kcIdx = 0; kcIdx < NK; ++kcIdx) {
        int kbase = kcIdx * 32 + grp * 16;   // never straddles a kk boundary
        int kk  = kbase >> CISHIFT;
        int ci0 = kbase & (CI - 1);
        int tin = 2 * t + kk - 1;            // >= -1; pad row holds zeros
        v16bf bfrag = *(const v16bf*)(inb + (ptrdiff_t)tin * CI + ci0);
        #pragma unroll
        for (int c = 0; c < NCOT; ++c) {
            v16bf afrag = *(const v16bf*)(wf + (((size_t)c * NK + kcIdx) * 32 + lane) * 16);
            acc[c] = __builtin_amdgcn_wmma_f32_16x16x32_bf16(false, afrag, false, bfrag,
                                                             (short)0, acc[c], false, false);
        }
    }
    // C/D layout: VGPR r holds (M = r + 8*grp, N = mn); M -> co, N -> t
    #pragma unroll
    for (int c = 0; c < NCOT; ++c) {
        int cob = c * 16 + 8 * grp;
        v8bf o;
        #pragma unroll
        for (int r = 0; r < 8; ++r) {
            int co = cob + r;
            float s = gam[co] * rsqrtf(var[co] + BN_EPS);
            float v = acc[c][r] * s + (bet[co] - mu[co] * s);
            o[r] = (bf16)fmaxf(v, 0.f);
        }
        *(v8bf*)(out + ((size_t)b * (LOUT + 1) + 1 + t) * CO + cob) = o;
    }
}

// ---------------------------------------------------------------------------
// GRU weight pre-swizzle: frag-ready bf16 B-fragments.
// Bf[gemm][nt][kc][lane][j] = W[col*128 + k], col = nt*16+mn, k = kc*32+(j<8 ? grp*8+j : 16+grp*8+j-8)
// ---------------------------------------------------------------------------
__global__ __launch_bounds__(256)
void prep_gru(const float* __restrict__ wih, const float* __restrict__ whh,
              bf16* __restrict__ bfw)
{
    int i = blockIdx.x * blockDim.x + threadIdx.x;
    if (i >= 2 * 24 * 4 * 32 * 16) return;
    int j    = i & 15;
    int lane = (i >> 4) & 31;
    int kc   = (i >> 9) & 3;
    int nt   = (i >> 11) % 24;
    int gemm = (i >> 11) / 24;
    int grp = lane >> 4, mn = lane & 15;
    int k   = kc * 32 + ((j < 8) ? (grp * 8 + j) : (16 + grp * 8 + (j - 8)));
    int col = nt * 16 + mn;
    const float* W = gemm ? whh : wih;
    bfw[i] = (bf16)W[(size_t)col * 128 + k];
}

// ---------------------------------------------------------------------------
// GRU: T=512 sequential steps; 2 blocks x 16 batch rows, 4 waves each.
// A-fragments hoisted per gemm per step (2x b128 each); B from frag-ready L2.
// Gates + h carry LDS-resident.
// act layout: [b][1+512][128] bf16 (padded layer-6 output, channels-last).
// ---------------------------------------------------------------------------
__global__ __launch_bounds__(128)
void gru_kernel(const bf16* __restrict__ act, const bf16* __restrict__ bfw,
                const float* __restrict__ bih, const float* __restrict__ bhh,
                float* __restrict__ zout)
{
    __shared__ float gi[16 * 384];
    __shared__ float gh[16 * 384];
    __shared__ float hbuf[16 * 128];
    __shared__ __align__(32) bf16 hb16[16 * 128];

    int tid  = threadIdx.x;
    int lane = tid & 31, wave = tid >> 5;
    int grp  = lane >> 4, mn = lane & 15;
    int b0   = blockIdx.x * 16;

    for (int i = tid; i < 16 * 128; i += 128) { hbuf[i] = 0.f; hb16[i] = (bf16)0.f; }
    __syncthreads();

    for (int t = 0; t < 512; ++t) {
        const bf16* xrow = act + ((size_t)(b0 + mn) * 513 + 1 + t) * 128;
        for (int sel = 0; sel < 2; ++sel) {
            // A-fragments for all 4 K-chunks (two contiguous 8-runs each)
            v16bf a[4];
            #pragma unroll
            for (int kci = 0; kci < 4; ++kci) {
                int kc = kci * 32;
                v8bf lo, hi;
                if (sel == 0) {
                    const v8bf* pa = (const v8bf*)(xrow + kc + grp * 8);
                    lo = pa[0]; hi = pa[2];
                } else {
                    const v8bf* ph = (const v8bf*)(&hb16[mn * 128 + kc + grp * 8]);
                    lo = ph[0]; hi = ph[2];
                }
                #pragma unroll
                for (int j = 0; j < 8; ++j) { a[kci][j] = lo[j]; a[kci][j + 8] = hi[j]; }
            }
            float* dst = sel ? gh : gi;
            const bf16* wsel = bfw + (size_t)sel * (24 * 4 * 32 * 16);
            for (int nt = wave; nt < 24; nt += 4) {
                v8f acc = {};
                #pragma unroll
                for (int kci = 0; kci < 4; ++kci) {
                    v16bf bfrag = *(const v16bf*)(wsel + (((size_t)nt * 4 + kci) * 32 + lane) * 16);
                    acc = __builtin_amdgcn_wmma_f32_16x16x32_bf16(false, a[kci], false, bfrag,
                                                                  (short)0, acc, false, false);
                }
                #pragma unroll
                for (int r = 0; r < 8; ++r)
                    dst[(r + 8 * grp) * 384 + nt * 16 + mn] = acc[r];
            }
        }
        __syncthreads();
        for (int e = tid; e < 16 * 128; e += 128) {
            int row = e >> 7, j = e & 127;
            float gir = gi[row * 384 + j]       + bih[j];
            float giz = gi[row * 384 + 128 + j] + bih[128 + j];
            float gin = gi[row * 384 + 256 + j] + bih[256 + j];
            float ghr = gh[row * 384 + j]       + bhh[j];
            float ghz = gh[row * 384 + 128 + j] + bhh[128 + j];
            float ghn = gh[row * 384 + 256 + j] + bhh[256 + j];
            float r = 1.f / (1.f + expf(-(gir + ghr)));
            float z = 1.f / (1.f + expf(-(giz + ghz)));
            float n = tanhf(gin + r * ghn);
            float hp = hbuf[e];
            float hn = (1.f - z) * n + z * hp;
            hbuf[e] = hn;
            hb16[e] = (bf16)hn;
        }
        __syncthreads();
    }
    for (int e = tid; e < 16 * 128; e += 128) {
        int row = e >> 7, j = e & 127;
        zout[(size_t)(b0 + row) * 128 + j] = hbuf[e];
    }
}

// ---------------------------------------------------------------------------
// 2-stage residual VQ (codebooks 7x128). One wave, lane = batch row.
// ---------------------------------------------------------------------------
__global__ __launch_bounds__(32)
void vq_kernel(const float* __restrict__ z,
               const float* __restrict__ cb0, const float* __restrict__ cb1,
               float* __restrict__ evec, float* __restrict__ loss_out,
               int* __restrict__ idx_out)
{
    __shared__ float zr[32 * 128];
    __shared__ float red[32];
    __shared__ int cnt[8];
    int b = threadIdx.x;
    for (int d = 0; d < 128; ++d) zr[b * 128 + d] = z[b * 128 + d];
    float loss = 0.f;
    for (int s = 0; s < 2; ++s) {
        const float* cb = s ? cb1 : cb0;
        float bestd = 3.4e38f; int best = 0;
        for (int e = 0; e < 7; ++e) {
            float dsum = 0.f;
            for (int d = 0; d < 128; ++d) {
                float df = zr[b * 128 + d] - cb[e * 128 + d];
                dsum += df * df;
            }
            if (dsum < bestd) { bestd = dsum; best = e; }
        }
        if (s == 0) idx_out[b] = best;
        for (int d = 0; d < 128; ++d) {
            float q = cb[best * 128 + d];
            evec[b * 256 + s * 128 + d] = q;
            zr[b * 128 + d] -= q;
        }
        red[b] = bestd;
        if (b < 8) cnt[b] = 0;
        __syncthreads();
        atomicAdd(&cnt[best], 1);
        __syncthreads();
        if (b == 0) {
            float tot = 0.f;
            for (int i = 0; i < 32; ++i) tot += red[i];
            float mse = tot / (32.f * 128.f);
            float usage = 0.f;
            for (int e = 0; e < 7; ++e) {
                float em = (float)cnt[e] / 32.f;
                usage -= em * logf(em + 1e-10f);
            }
            loss += 1.4f * mse + 0.01f * usage;
        }
        __syncthreads();
    }
    if (b == 0) *loss_out = loss;
}

// ---------------------------------------------------------------------------
// Output assembly: style(32x768) | loss(1) | idx(32) | e1,e2,e3,e1+e2+e3
// ---------------------------------------------------------------------------
__global__ __launch_bounds__(256)
void finalize_kernel(const float* __restrict__ evec, const float* __restrict__ lossbuf,
                     const int* __restrict__ idxbuf, float* __restrict__ out)
{
    int i = blockIdx.x * blockDim.x + threadIdx.x;
    if (i < 32 * 256) {
        int b = i >> 8, d = i & 255;
        float v1 = evec[0 * 8192 + i];
        float v2 = evec[1 * 8192 + i];
        float v3 = evec[2 * 8192 + i];
        out[b * 768 + d]       = v1;
        out[b * 768 + 256 + d] = v2;
        out[b * 768 + 512 + d] = v3;
        float* cbout = out + 32 * 768 + 1 + 32;
        cbout[i]            = v1;
        cbout[8192 + i]     = v2;
        cbout[2 * 8192 + i] = v3;
        cbout[3 * 8192 + i] = v1 + v2 + v3;
    }
    if (i == 0)  out[32 * 768] = lossbuf[0] + lossbuf[1] + lossbuf[2];
    if (i < 32)  out[32 * 768 + 1 + i] =
        (float)(idxbuf[i] + idxbuf[32 + i] + idxbuf[64 + i]);
}

// ---------------------------------------------------------------------------
// Host launcher. Input flattening (setup_inputs insertion order; enc dicts in
// their own insertion order -> 34 leaves each):
//   0: ref_embs(unused) 1..3: p/d/e_targets
//   enc base = 4 + 34*enc : conv_w[0..5], bn_gamma[0..5], bn_beta[0..5],
//                           bn_mean[0..5], bn_var[0..5], w_ih, w_hh, b_ih, b_hh
//   106..111: cb_p0, cb_p1, cb_d0, cb_d1, cb_e0, cb_e1
// ---------------------------------------------------------------------------
extern "C" void kernel_launch(void* const* d_in, const int* in_sizes, int n_in,
                              void* d_out, int out_size, void* d_ws, size_t ws_size,
                              hipStream_t stream)
{
    (void)in_sizes; (void)n_in; (void)out_size; (void)ws_size;

    char* w8 = (char*)d_ws;
    bf16*  actA   = (bf16*)w8;                                   // 33,558,528 B
    bf16*  actB   = (bf16*)(w8 + (size_t)33562624);              // 16,779,264 B
    float* zb     = (float*)(w8 + (size_t)50348032);             // 49,152 B
    float* evec   = (float*)(w8 + (size_t)50397184);             // 98,304 B
    float* lossb  = (float*)(w8 + (size_t)50495488);             // 256 B
    int*   idxb   = (int*)  (w8 + (size_t)50495744);             // 512 B
    bf16*  convWf = (bf16*)(w8 + (size_t)50496256);              // 98,304 B
    bf16*  gruWf  = (bf16*)(w8 + (size_t)50594560);              // 196,608 B

    static const int ENC_BASE[3] = {4, 38, 72};
    static const int TGT[3]      = {1, 2, 3};
    static const int CBI[3]      = {106, 108, 110};
    struct Cfg { int Ci, Co, Lin, Lout, ciShift; };
    static const Cfg cfg[6] = {
        {1, 32, 32768, 16384, 0}, {32, 32, 16384, 8192, 5}, {32, 64, 8192, 4096, 5},
        {64, 64, 4096, 2048, 6},  {64, 128, 2048, 1024, 6}, {128, 128, 1024, 512, 7}};

    for (int enc = 0; enc < 3; ++enc) {
        int base = ENC_BASE[enc];
        const float* x = (const float*)d_in[TGT[enc]];

        // Layer 1 (padded channels-last out in actA)
        {
            const float* w   = (const float*)d_in[base + 0];
            const float* gam = (const float*)d_in[base + 6];
            const float* bet = (const float*)d_in[base + 12];
            const float* mu  = (const float*)d_in[base + 18];
            const float* var = (const float*)d_in[base + 24];
            zero_pad<<<(32 * 32 + 255) / 256, 256, 0, stream>>>(
                actA, 32, (cfg[0].Lout + 1) * 32);
            int total = 32 * 32 * 16384;
            conv1_kernel<<<(total + 255) / 256, 256, 0, stream>>>(
                x, w, gam, bet, mu, var, actA, cfg[0].Lin, cfg[0].Lout);
        }
        // Layers 2..6 (WMMA implicit-GEMM, ping-pong actA/actB)
        bf16* bufs[2] = {actA, actB};
        for (int L = 1; L < 6; ++L) {
            const float* w   = (const float*)d_in[base + 0 + L];
            const float* gam = (const float*)d_in[base + 6 + L];
            const float* bet = (const float*)d_in[base + 12 + L];
            const float* mu  = (const float*)d_in[base + 18 + L];
            const float* var = (const float*)d_in[base + 24 + L];
            int nk = (3 * cfg[L].Ci) >> 5;
            int wtotal = cfg[L].Co * 3 * cfg[L].Ci;
            prep_convw<<<(wtotal + 255) / 256, 256, 0, stream>>>(
                w, convWf, cfg[L].Ci, cfg[L].Co, cfg[L].ciShift, nk);
            bf16* ib = bufs[(L - 1) & 1];
            bf16* ob = bufs[L & 1];
            zero_pad<<<(32 * cfg[L].Co + 255) / 256, 256, 0, stream>>>(
                ob, cfg[L].Co, (cfg[L].Lout + 1) * cfg[L].Co);
            dim3 grid(cfg[L].Lout / 16, 32);
            switch (L) {
            case 1: convw_kernel<32, 32, 16384, 8192><<<grid, 32, 0, stream>>>(
                        ib, convWf, gam, bet, mu, var, ob); break;
            case 2: convw_kernel<32, 64, 8192, 4096><<<grid, 32, 0, stream>>>(
                        ib, convWf, gam, bet, mu, var, ob); break;
            case 3: convw_kernel<64, 64, 4096, 2048><<<grid, 32, 0, stream>>>(
                        ib, convWf, gam, bet, mu, var, ob); break;
            case 4: convw_kernel<64, 128, 2048, 1024><<<grid, 32, 0, stream>>>(
                        ib, convWf, gam, bet, mu, var, ob); break;
            case 5: convw_kernel<128, 128, 1024, 512><<<grid, 32, 0, stream>>>(
                        ib, convWf, gam, bet, mu, var, ob); break;
            }
        }
        // GRU over layer-6 output (actB, [b][1+512][128])
        {
            const float* wih = (const float*)d_in[base + 30];
            const float* whh = (const float*)d_in[base + 31];
            const float* bih = (const float*)d_in[base + 32];
            const float* bhh = (const float*)d_in[base + 33];
            prep_gru<<<(98304 + 255) / 256, 256, 0, stream>>>(wih, whh, gruWf);
            gru_kernel<<<2, 128, 0, stream>>>(actB, gruWf, bih, bhh,
                                              zb + (size_t)enc * 4096);
        }
        // RVQ2
        {
            const float* cb0 = (const float*)d_in[CBI[enc]];
            const float* cb1 = (const float*)d_in[CBI[enc] + 1];
            vq_kernel<<<1, 32, 0, stream>>>(zb + (size_t)enc * 4096, cb0, cb1,
                                            evec + (size_t)enc * 8192,
                                            lossb + enc, idxb + enc * 32);
        }
    }
    finalize_kernel<<<32, 256, 0, stream>>>(evec, lossb, idxb, (float*)d_out);
}
